// MLP_self_74981539053834
// MI455X (gfx1250) — matrix operand order repaired and checked by reading, not compiled, not run on hardware
//
#include <hip/hip_runtime.h>
#include <stdint.h>

// ---------------------------------------------------------------------------
// MLP: out = mean_rows( relu(relu(x*mask @ W1^T + b1) @ W2^T + b2) @ W3^T + b3 )
// n=4096, d_in=1024, d_h=4096, d_out=1024.  bf16 WMMA GEMMs, f32 accumulate.
// Last (linear) layer folded through the row-mean: colsum(h2) -> matvec.
// GEMM staging uses gfx1250 async global->LDS loads with double buffering.
// ---------------------------------------------------------------------------

typedef __attribute__((ext_vector_type(16))) __bf16 v16bf;
typedef __attribute__((ext_vector_type(8)))  float  v8f;

#define N_ROWS 4096
#define D_IN   1024
#define D_H    4096
#define D_OUT  1024

// ---------------- helpers ----------------

__device__ __forceinline__ unsigned short f32_to_bf16(float f) {
  uint32_t u = __float_as_uint(f);
  if ((u & 0x7fffffffu) > 0x7f800000u) return (unsigned short)((u >> 16) | 0x40); // qNaN
  u += 0x7fffu + ((u >> 16) & 1u);  // round-to-nearest-even
  return (unsigned short)(u >> 16);
}

__device__ __forceinline__ uint64_t splitmix64(uint64_t z) {
  z ^= z >> 30; z *= 0xBF58476D1CE4E5B9ull;
  z ^= z >> 27; z *= 0x94D049BB133111EBull;
  z ^= z >> 31; return z;
}

// Async global->LDS copy of 16 bytes (ASYNCcnt-tracked, no VGPR data path).
// Inline asm keeps this portable across ROCm-7.2 / upstream toolchains whose
// builtin arities differ.
__device__ __forceinline__ void async_cp16(void* lds_dst, const void* gsrc) {
  asm volatile("global_load_async_to_lds_b128 %0, %1, off"
               :
               : "v"((uint32_t)(uintptr_t)lds_dst),
                 "v"((unsigned long long)(uintptr_t)gsrc)
               : "memory");
}

// ---------------- dedup mask ----------------

// Commutative (sum) hash per row -> deterministic under any reduction order.
__global__ void row_hash_kernel(const uint32_t* __restrict__ x,
                                uint64_t* __restrict__ hashes, int d) {
  __shared__ uint64_t sh[256];
  int row = blockIdx.x;
  const uint32_t* p = x + (size_t)row * d;
  uint64_t h = 0;
  for (int k = threadIdx.x; k < d; k += 256)
    h += splitmix64(((uint64_t)p[k] << 1) ^ (0x9E3779B97F4A7C15ull * (uint64_t)(k + 1)));
  sh[threadIdx.x] = h; __syncthreads();
  for (int s = 128; s > 0; s >>= 1) {
    if (threadIdx.x < s) sh[threadIdx.x] += sh[threadIdx.x + s];
    __syncthreads();
  }
  if (threadIdx.x == 0) hashes[row] = sh[0];
}

// mask[i] = 0 iff some j<i has identical row (hash prefilter, full check on hit)
__global__ void dedup_mask_kernel(const float* __restrict__ x,
                                  const uint64_t* __restrict__ hashes,
                                  float* __restrict__ mask, int d) {
  __shared__ int flag;
  int i = blockIdx.x;
  if (threadIdx.x == 0) flag = 0;
  __syncthreads();
  uint64_t hi = hashes[i];
  for (int j = threadIdx.x; j < i; j += 256) {
    if (hashes[j] == hi) {
      const float* a = x + (size_t)i * d;
      const float* b = x + (size_t)j * d;
      bool eq = true;
      for (int k = 0; k < d && eq; ++k) eq = (a[k] == b[k]);
      if (eq) atomicOr(&flag, 1);
    }
  }
  __syncthreads();
  if (threadIdx.x == 0) mask[i] = flag ? 0.0f : 1.0f;
}

// ---------------- conversion / init ----------------

__global__ void f32_to_bf16_kernel(const float* __restrict__ src,
                                   unsigned short* __restrict__ dst,
                                   const float* __restrict__ rowscale, // nullable
                                   int cols, int total4) {
  int i4 = blockIdx.x * blockDim.x + threadIdx.x;
  if (i4 >= total4) return;
  int idx = i4 * 4;
  float4 v = *(const float4*)(src + idx);
  if (rowscale) {
    float m = rowscale[idx / cols];  // cols % 4 == 0 -> same row for all 4
    v.x *= m; v.y *= m; v.z *= m; v.w *= m;
  }
  ushort4 o;
  o.x = f32_to_bf16(v.x); o.y = f32_to_bf16(v.y);
  o.z = f32_to_bf16(v.z); o.w = f32_to_bf16(v.w);
  *(ushort4*)(dst + idx) = o;
}

__global__ void zero_f32_kernel(float* __restrict__ p, int n) {
  int i = blockIdx.x * blockDim.x + threadIdx.x;
  if (i < n) p[i] = 0.0f;
}

// ---------------- WMMA GEMM:  C = relu(A @ W^T + b) ----------------
// A: [M,K] bf16 row-major, W: [N,K] bf16 row-major (so B=W^T is K-major per column).
// Block tile 128x128, BK=64, 8 waves (2x4), wave tile 64x32 (4x2 wmma accums).
// Double-buffered LDS filled by ASYNCcnt-tracked global_load_async_to_lds_b128.

#define LDT  72                 // padded LDS stride (halfwords): 144B rows
#define SBUF (128 * LDT)        // halfwords per matrix tile
#define LDS_BYTES (2 * 2 * SBUF * 2)  // 2 buffers x (A+B) x 2B = 73728 B

template<bool STORE, bool COLSUM>
__global__ __launch_bounds__(256) void gemm_relu_bf16(
    const unsigned short* __restrict__ A, const unsigned short* __restrict__ W,
    const float* __restrict__ bias, unsigned short* __restrict__ Out,
    float* __restrict__ colsum, int M, int N, int K) {

  extern __shared__ __align__(16) unsigned short smem[];  // [2][2][SBUF]

  const int tid  = threadIdx.x;
  const int wid  = tid >> 5;
  const int lane = tid & 31;
  const int lr   = lane & 15;     // lane row within half-wave
  const int hf   = lane >> 4;     // 0: lanes 0-15, 1: lanes 16-31
  const int wm   = wid & 1;       // wave M position (2)
  const int wn   = wid >> 1;      // wave N position (4)
  const int m0   = blockIdx.y * 128;
  const int n0   = blockIdx.x * 128;

  // Per-thread staging coords: 8 x 16B async copies per tile (4 for A, 4 for B).
  const int srow = tid >> 3;            // four of 128 rows per thread (srow + 32*it, it = 0..3)
  const int scc  = (tid & 7) * 8;       // halfword column within 64-wide tile

  // Issue the 8 async loads for K-tile k0 into buffer b.
  auto stage = [&](int k0, int b) {
    unsigned short* dA = smem + b * (2 * SBUF);
    unsigned short* dB = dA + SBUF;
#pragma unroll
    for (int it = 0; it < 4; ++it) {
      int row = srow + it * 32;
      async_cp16(dA + row * LDT + scc, A + (size_t)(m0 + row) * K + k0 + scc);
      async_cp16(dB + row * LDT + scc, W + (size_t)(n0 + row) * K + k0 + scc);
    }
  };

  v8f acc[4][2];
  v8f vzero = {};
#pragma unroll
  for (int t = 0; t < 4; ++t)
#pragma unroll
    for (int u = 0; u < 2; ++u) acc[t][u] = vzero;

  stage(0, 0);

  int buf = 0;
  for (int k0 = 0; k0 < K; k0 += 64, buf ^= 1) {
    const bool more = (k0 + 64) < K;
    if (more) stage(k0 + 64, buf ^ 1);  // prefetch next tile into other buffer

    // Async loads retire in order: <=8 outstanding => current tile is in LDS.
    if (more) asm volatile("s_wait_asynccnt 0x8" ::: "memory");
    else      asm volatile("s_wait_asynccnt 0x0" ::: "memory");
    __syncthreads();

    const unsigned short* sA = smem + buf * (2 * SBUF);
    const unsigned short* sB = sA + SBUF;

#pragma unroll
    for (int ks = 0; ks < 2; ++ks) {
      const int kk = ks * 32;
      // A fragments (16x32): lane lr(+16*hf) row; K = {8h..8h+7, 16+8h..16+8h+7}
      v16bf af[4];
#pragma unroll
      for (int t = 0; t < 4; ++t) {
        const unsigned short* p = &sA[(wm * 64 + t * 16 + lr) * LDT + kk];
        __builtin_memcpy((char*)&af[t],      p + 8 * hf,      16);
        __builtin_memcpy((char*)&af[t] + 16, p + 16 + 8 * hf, 16);
      }
      // B fragments (32x16): lane lr holds W[n, kk + {8h..8h+7, 16+8h..16+8h+7}]
      v16bf bf[2];
#pragma unroll
      for (int u = 0; u < 2; ++u) {
        const unsigned short* p = &sB[(wn * 32 + u * 16 + lr) * LDT + kk];
        __builtin_memcpy((char*)&bf[u],      p + 8 * hf,      16);
        __builtin_memcpy((char*)&bf[u] + 16, p + 16 + 8 * hf, 16);
      }
#pragma unroll
      for (int t = 0; t < 4; ++t)
#pragma unroll
        for (int u = 0; u < 2; ++u)
          acc[t][u] = __builtin_amdgcn_wmma_f32_16x16x32_bf16(
              false, af[t], false, bf[u], (short)0, acc[t][u], false, false);
    }
    __syncthreads();  // all waves done reading buf before it is refilled
  }

  // Epilogue: bias + relu; store bf16 and/or accumulate column sums.
  float csum[2] = {0.0f, 0.0f};
#pragma unroll
  for (int u = 0; u < 2; ++u) {
    const int gn = n0 + wn * 32 + u * 16 + lr;
    const float bv = bias[gn];
#pragma unroll
    for (int t = 0; t < 4; ++t) {
#pragma unroll
      for (int r = 0; r < 8; ++r) {
        const int gm = m0 + wm * 64 + t * 16 + 8 * hf + r;  // C layout: lanes16-31 -> M+8
        float v = acc[t][u][r] + bv;
        v = fmaxf(v, 0.0f);
        if (STORE)  Out[(size_t)gm * N + gn] = f32_to_bf16(v);
        if (COLSUM) csum[u] += v;
      }
    }
  }
  if (COLSUM) {
#pragma unroll
    for (int u = 0; u < 2; ++u)
      atomicAdd(&colsum[n0 + wn * 32 + u * 16 + lr], csum[u]);
  }
}

// ---------------- final: out = b3 + (1/n) * W3 @ colsum ----------------

__global__ void mean_matvec_kernel(const float* __restrict__ W3,
                                   const float* __restrict__ b3,
                                   const float* __restrict__ colsum,
                                   float* __restrict__ out, int H, float inv_n) {
  __shared__ float sh[256];
  int o = blockIdx.x;
  const float* w = W3 + (size_t)o * H;
  float acc = 0.0f;
  for (int k = threadIdx.x; k < H; k += 256) acc += w[k] * colsum[k];
  sh[threadIdx.x] = acc; __syncthreads();
  for (int s = 128; s > 0; s >>= 1) {
    if (threadIdx.x < s) sh[threadIdx.x] += sh[threadIdx.x + s];
    __syncthreads();
  }
  if (threadIdx.x == 0) out[o] = b3[o] + inv_n * sh[0];
}

// ---------------- host launcher ----------------

extern "C" void kernel_launch(void* const* d_in, const int* in_sizes, int n_in,
                              void* d_out, int out_size, void* d_ws, size_t ws_size,
                              hipStream_t stream) {
  const float* x  = (const float*)d_in[0];
  const float* W1 = (const float*)d_in[1];
  const float* b1 = (const float*)d_in[2];
  const float* W2 = (const float*)d_in[3];
  const float* b2 = (const float*)d_in[4];
  const float* W3 = (const float*)d_in[5];
  const float* b3 = (const float*)d_in[6];

  char* ws = (char*)d_ws;
  size_t off = 0;
  unsigned short* xm  = (unsigned short*)(ws + off); off += (size_t)N_ROWS * D_IN * 2; // 8 MiB
  unsigned short* w1b = (unsigned short*)(ws + off); off += (size_t)D_H * D_IN * 2;    // 8 MiB
  unsigned short* w2b = (unsigned short*)(ws + off); off += (size_t)D_H * D_H * 2;     // 32 MiB
  unsigned short* h1b = (unsigned short*)(ws + off); off += (size_t)N_ROWS * D_H * 2;  // 32 MiB
  uint64_t* hashes    = (uint64_t*)(ws + off);       off += (size_t)N_ROWS * 8;
  float* mask         = (float*)(ws + off);          off += (size_t)N_ROWS * 4;
  float* colsum       = (float*)(ws + off);          off += (size_t)D_H * 4;
  (void)ws_size; (void)in_sizes; (void)n_in; (void)out_size;

  // 1) dedup mask
  row_hash_kernel<<<N_ROWS, 256, 0, stream>>>((const uint32_t*)x, hashes, D_IN);
  dedup_mask_kernel<<<N_ROWS, 256, 0, stream>>>(x, hashes, mask, D_IN);

  // 2) bf16 staging (x masked in the same pass)
  {
    int t4 = N_ROWS * D_IN / 4;
    f32_to_bf16_kernel<<<(t4 + 255) / 256, 256, 0, stream>>>(x, xm, mask, D_IN, t4);
  }
  {
    int t4 = D_H * D_IN / 4;
    f32_to_bf16_kernel<<<(t4 + 255) / 256, 256, 0, stream>>>(W1, w1b, nullptr, D_IN, t4);
  }
  {
    int t4 = D_H * D_H / 4;
    f32_to_bf16_kernel<<<(t4 + 255) / 256, 256, 0, stream>>>(W2, w2b, nullptr, D_H, t4);
  }
  zero_f32_kernel<<<(D_H + 255) / 256, 256, 0, stream>>>(colsum, D_H);

  // 3) GEMM1: h1 = relu(xm @ W1^T + b1)   -> bf16
  gemm_relu_bf16<true, false>
      <<<dim3(D_H / 128, N_ROWS / 128), 256, LDS_BYTES, stream>>>(
          xm, w1b, b1, h1b, nullptr, N_ROWS, D_H, D_IN);

  // 4) GEMM2: colsum = sum_rows relu(h1 @ W2^T + b2)   (h2 never materialized)
  gemm_relu_bf16<false, true>
      <<<dim3(D_H / 128, N_ROWS / 128), 256, LDS_BYTES, stream>>>(
          h1b, w2b, b2, nullptr, colsum, N_ROWS, D_H, D_H);

  // 5) out = b3 + W3 @ (colsum / n)   (linear layer commutes with the mean)
  mean_matvec_kernel<<<D_OUT, 256, 0, stream>>>(W3, b3, colsum, (float*)d_out,
                                                D_H, 1.0f / (float)N_ROWS);
}
